// TensorBase_566935683512
// MI455X (gfx1250) — matrix operand order repaired and checked by reading, not compiled, not run on hardware
//
#include <hip/hip_runtime.h>

typedef __attribute__((ext_vector_type(16))) _Float16 v16h;
typedef __attribute__((ext_vector_type(8)))  _Float16 v8h;
typedef __attribute__((ext_vector_type(8)))  float    v8f;

#define S_SAMPLES 384
#define NEAR_T 2.0f
#define FAR_T 6.0f
#define DENSITY_SHIFT (-10.0f)
#define DIST_SCALE 25.0f
#define VOL 2097152   // 128^3
#define FSTRIDE 136   // feature row stride (halves): 272B, 16B aligned, bank-staggered

// LDS partition (dynamic shared memory)
#define WGT_HALVES  (68 * 512)                 // 68 fragments * 512 halves
#define WGT_BYTES   (WGT_HALVES * 2)           // 69632
#define FEAT_BYTES  (8 * 16 * FSTRIDE * 2)     // 34816
#define SW_BYTES    (S_SAMPLES * 4)            // 1536
#define SMEM_BYTES  (WGT_BYTES + FEAT_BYTES + SW_BYTES + 16)

union H16 { v16h v; v8h h2[2]; _Float16 e[16]; };
union F8  { v8f  v; float e[8]; };

__device__ __forceinline__ int clampi(int v, int lo, int hi) {
  return v < lo ? lo : (v > hi ? hi : v);
}

struct TriW {
  int i000,i001,i010,i011,i100,i101,i110,i111;
  float w000,w001,w010,w011,w100,w101,w110,w111;
};

// coords = clip(pts * 2/3, -1, 1); align_corners grid coords in [0,127]
__device__ __forceinline__ TriW tri_setup(float px, float py, float pz) {
  float cx = fminf(fmaxf(px * (2.0f/3.0f), -1.0f), 1.0f);
  float cy = fminf(fmaxf(py * (2.0f/3.0f), -1.0f), 1.0f);
  float cz = fminf(fmaxf(pz * (2.0f/3.0f), -1.0f), 1.0f);
  float fx = (cx + 1.0f) * 63.5f;
  float fy = (cy + 1.0f) * 63.5f;
  float fz = (cz + 1.0f) * 63.5f;
  float x0f = floorf(fx), y0f = floorf(fy), z0f = floorf(fz);
  int x0 = clampi((int)x0f, 0, 127);
  int y0 = clampi((int)y0f, 0, 127);
  int z0 = clampi((int)z0f, 0, 127);
  int x1 = x0 + 1 > 127 ? 127 : x0 + 1;
  int y1 = y0 + 1 > 127 ? 127 : y0 + 1;
  int z1 = z0 + 1 > 127 ? 127 : z0 + 1;
  float wx = fx - x0f, wy = fy - y0f, wz = fz - z0f;
  float ux = 1.0f - wx, uy = 1.0f - wy, uz = 1.0f - wz;
  TriW t;
  t.i000 = (z0*128 + y0)*128 + x0;  t.i001 = (z0*128 + y0)*128 + x1;
  t.i010 = (z0*128 + y1)*128 + x0;  t.i011 = (z0*128 + y1)*128 + x1;
  t.i100 = (z1*128 + y0)*128 + x0;  t.i101 = (z1*128 + y0)*128 + x1;
  t.i110 = (z1*128 + y1)*128 + x0;  t.i111 = (z1*128 + y1)*128 + x1;
  t.w000 = ux*uy*uz; t.w001 = wx*uy*uz; t.w010 = ux*wy*uz; t.w011 = wx*wy*uz;
  t.w100 = ux*uy*wz; t.w101 = wx*uy*wz; t.w110 = ux*wy*wz; t.w111 = wx*wy*wz;
  return t;
}

// ---------------------------------------------------------------------------
// Weight conversion: f32 row-major [K][N] -> f16 WMMA B fragments, packed so
// each lane's 16 halves are contiguous (32B per lane).
// B k order used here (contiguous, as A): lane n<16 -> col n, K=kbase+0..15 (half h -> K+h);
// lane n>=16 -> col n-16, K=kbase+16..31.
// Fragment index: 0..31 = W1 (kt*8+nt), 32..63 = W2, 64..67 = W3 (kt).
// ---------------------------------------------------------------------------
__global__ void wconvert(const float* __restrict__ W1,
                         const float* __restrict__ W2,
                         const float* __restrict__ W3,
                         _Float16* __restrict__ pack) {
  int frag = blockIdx.x;        // 0..67
  int t    = threadIdx.x;       // 0..511
  int lane = t >> 4;
  int h    = t & 15;
  const float* W; int K, Nn, NT, stride, fb;
  if (frag < 32)      { W = W1; K = 102; Nn = 128; NT = 8; stride = 128; fb = frag; }
  else if (frag < 64) { W = W2; K = 128; Nn = 128; NT = 8; stride = 128; fb = frag - 32; }
  else                { W = W3; K = 128; Nn = 3;   NT = 1; stride = 3;   fb = frag - 64; }
  int kt  = fb / NT;
  int nt  = fb % NT;
  int col = nt * 16 + (lane & 15);
  int k   = kt * 32 + ((lane & 16) ? 16 : 0) + h;
  float v = (k < K && col < Nn) ? W[k * stride + col] : 0.0f;
  pack[(size_t)(frag * 32 + lane) * 16 + h] = (_Float16)v;
}

// B fragment from LDS-staged weights: 32B contiguous per lane (2x ds_load_b128)
__device__ __forceinline__ v16h load_bfrag(const _Float16* wgt, int frag, int lane) {
  const _Float16* p = wgt + (((unsigned)(frag * 32 + lane)) << 4);
  H16 b;
  b.h2[0] = *(const v8h*)(p);
  b.h2[1] = *(const v8h*)(p + 8);
  return b.v;
}

// A k order used here (contiguous, as B): lane holds row m = lane&15.
// lane<16: K = kt*32 + 0..15 (halves 0..15)
// lane>=16: K = kt*32 + 16..31.
__device__ __forceinline__ v16h load_afrag(const _Float16* frow, int kt, int lane) {
  int kb = (lane & 16) ? 16 : 0;
  H16 a;
  a.h2[0] = *(const v8h*)(frow + kt * 32 + kb);
  a.h2[1] = *(const v8h*)(frow + kt * 32 + kb + 8);
  return a.v;
}

// ---------------------------------------------------------------------------
// One block (8 waves) per ray. Each wave shades 48 samples as 3 tiles of 16.
// All packed weights staged into LDS once per block.
// ---------------------------------------------------------------------------
__global__ __launch_bounds__(256)
void render_kernel(const float* __restrict__ rays_o,
                   const float* __restrict__ rays_d,
                   const float* __restrict__ dens,
                   const float* __restrict__ app,
                   const float* __restrict__ b1,
                   const float* __restrict__ b2,
                   const float* __restrict__ b3,
                   const _Float16* __restrict__ pack,
                   float* __restrict__ out) {
  extern __shared__ __align__(128) char smem[];
  _Float16* s_wgt  = (_Float16*)smem;                              // 69632 B
  _Float16* s_feat = (_Float16*)(smem + WGT_BYTES);                // 34816 B
  float*    s_w    = (float*)(smem + WGT_BYTES + FEAT_BYTES);      // 1536 B
  float*    s_rgb  = (float*)(smem + WGT_BYTES + FEAT_BYTES + SW_BYTES);
  float*    s_acc  = s_rgb + 3;

  const int ray  = blockIdx.x;
  const int tid  = threadIdx.x;
  const int lane = tid & 31;
  const int wave = tid >> 5;

  // ---------------- Stage weights into LDS ----------------
  {
    const uint4* src = (const uint4*)pack;
    uint4* dst = (uint4*)s_wgt;
    for (int i = tid; i < WGT_BYTES / 16; i += 256) dst[i] = src[i];
  }

  const float ox = rays_o[ray*3+0], oy = rays_o[ray*3+1], oz = rays_o[ray*3+2];
  const float dx = rays_d[ray*3+0], dy = rays_d[ray*3+1], dzv = rays_d[ray*3+2];
  const float dstep = (FAR_T - NEAR_T) / (float)(S_SAMPLES - 1);

  // ---------------- Phase 1: density -> alpha ----------------
  for (int s = tid; s < S_SAMPLES; s += 256) {
    float z  = NEAR_T + dstep * (float)s;
    float px = ox + dx * z, py = oy + dy * z, pz = oz + dzv * z;
    bool inb = (px >= -1.5f) && (px <= 1.5f) &&
               (py >= -1.5f) && (py <= 1.5f) &&
               (pz >= -1.5f) && (pz <= 1.5f);
    TriW t = tri_setup(px, py, pz);
    float feat = t.w000*dens[t.i000] + t.w001*dens[t.i001]
               + t.w010*dens[t.i010] + t.w011*dens[t.i011]
               + t.w100*dens[t.i100] + t.w101*dens[t.i101]
               + t.w110*dens[t.i110] + t.w111*dens[t.i111];
    float xs  = feat + DENSITY_SHIFT;
    float sig = (xs > 20.0f) ? xs : log1pf(__expf(xs));
    sig *= inb ? 1.0f : 0.0f;
    float dist = (s == S_SAMPLES - 1) ? 0.0f : dstep;
    s_w[s] = 1.0f - __expf(-sig * dist * DIST_SCALE);
  }
  if (tid < 3) s_rgb[tid] = 0.0f;
  __syncthreads();

  // ---------------- Phase 2: serial cumprod -> weights ----------------
  if (tid == 0) {
    float T = 1.0f, acc = 0.0f;
    for (int s = 0; s < S_SAMPLES; ++s) {
      float a = s_w[s];
      float w = a * T;
      s_w[s] = w;
      acc += w;
      T *= (1.0f - a + 1e-10f);
    }
    *s_acc = acc;
  }
  __syncthreads();

  // ---------------- Phase 3: shading (WMMA MLP) ----------------
  const _Float16* W1p = s_wgt;
  const _Float16* W2p = s_wgt + 32 * 512;
  const _Float16* W3p = s_wgt + 64 * 512;
  const int m   = lane & 15;            // sample row within tile
  const int mhi = (lane & 16) ? 8 : 0;  // D-matrix row offset for this half-wave
  const int col = m;                    // D-matrix column for this lane
  const bool hi = (lane & 16) != 0;

  for (int tt = 0; tt < 3; ++tt) {
    const int base = wave * 48 + tt * 16;
    const int s    = base + m;
    float z  = NEAR_T + dstep * (float)s;
    float px = ox + dx * z, py = oy + dy * z, pz = oz + dzv * z;
    _Float16* f = s_feat + (wave * 16 + m) * FSTRIDE;

    // ---- features: both half-waves gather ~half the app channels + one PE block
    {
      TriW t = tri_setup(px, py, pz);
      int c0 = hi ? 14 : 0;
      int c1 = hi ? 27 : 14;
      for (int c = c0; c < c1; ++c) {
        const float* a = app + (size_t)c * VOL;
        float v = t.w000*a[t.i000] + t.w001*a[t.i001]
                + t.w010*a[t.i010] + t.w011*a[t.i011]
                + t.w100*a[t.i100] + t.w101*a[t.i101]
                + t.w110*a[t.i110] + t.w111*a[t.i111];
        f[c] = (_Float16)v;
      }
      if (!hi) {
        // positional encoding of pts -> features 30..65
        float P[3] = {px, py, pz};
        for (int c = 0; c < 3; ++c) {
          float fb = 1.0f;
          for (int q = 0; q < 6; ++q) {
            float sn, cs;
            __sincosf(P[c] * fb, &sn, &cs);
            f[30 + c*6 + q] = (_Float16)sn;
            f[48 + c*6 + q] = (_Float16)cs;
            fb *= 2.0f;
          }
        }
      } else {
        // viewdirs + view PE -> features 27..29, 66..101, zero pad 102..127
        f[27] = (_Float16)dx; f[28] = (_Float16)dy; f[29] = (_Float16)dzv;
        float D[3] = {dx, dy, dzv};
        for (int c = 0; c < 3; ++c) {
          float fb = 1.0f;
          for (int q = 0; q < 6; ++q) {
            float sn, cs;
            __sincosf(D[c] * fb, &sn, &cs);
            f[66 + c*6 + q] = (_Float16)sn;
            f[84 + c*6 + q] = (_Float16)cs;
            fb *= 2.0f;
          }
        }
        for (int j = 102; j < 128; ++j) f[j] = (_Float16)0.0f;
      }
    }
    __syncthreads();

    // ---- layer 1: [16x128] x [128x128] ----
    v16h A0 = load_afrag(f, 0, lane);
    v16h A1 = load_afrag(f, 1, lane);
    v16h A2 = load_afrag(f, 2, lane);
    v16h A3 = load_afrag(f, 3, lane);
    F8 acc1[8];
#pragma unroll
    for (int nt = 0; nt < 8; ++nt) {
      float bb = b1[nt*16 + col];
#pragma unroll
      for (int v = 0; v < 8; ++v) acc1[nt].e[v] = bb;
      acc1[nt].v = __builtin_amdgcn_wmma_f32_16x16x32_f16(false, A0, false, load_bfrag(W1p, 0*8+nt, lane), (short)0, acc1[nt].v, false, false);
      acc1[nt].v = __builtin_amdgcn_wmma_f32_16x16x32_f16(false, A1, false, load_bfrag(W1p, 1*8+nt, lane), (short)0, acc1[nt].v, false, false);
      acc1[nt].v = __builtin_amdgcn_wmma_f32_16x16x32_f16(false, A2, false, load_bfrag(W1p, 2*8+nt, lane), (short)0, acc1[nt].v, false, false);
      acc1[nt].v = __builtin_amdgcn_wmma_f32_16x16x32_f16(false, A3, false, load_bfrag(W1p, 3*8+nt, lane), (short)0, acc1[nt].v, false, false);
    }
    __syncthreads();
#pragma unroll
    for (int nt = 0; nt < 8; ++nt)
#pragma unroll
      for (int v = 0; v < 8; ++v)
        s_feat[(wave * 16 + v + mhi) * FSTRIDE + nt*16 + col] =
            (_Float16)fmaxf(acc1[nt].e[v], 0.0f);
    __syncthreads();

    // ---- layer 2: [16x128] x [128x128] ----
    A0 = load_afrag(f, 0, lane);
    A1 = load_afrag(f, 1, lane);
    A2 = load_afrag(f, 2, lane);
    A3 = load_afrag(f, 3, lane);
    F8 acc2[8];
#pragma unroll
    for (int nt = 0; nt < 8; ++nt) {
      float bb = b2[nt*16 + col];
#pragma unroll
      for (int v = 0; v < 8; ++v) acc2[nt].e[v] = bb;
      acc2[nt].v = __builtin_amdgcn_wmma_f32_16x16x32_f16(false, A0, false, load_bfrag(W2p, 0*8+nt, lane), (short)0, acc2[nt].v, false, false);
      acc2[nt].v = __builtin_amdgcn_wmma_f32_16x16x32_f16(false, A1, false, load_bfrag(W2p, 1*8+nt, lane), (short)0, acc2[nt].v, false, false);
      acc2[nt].v = __builtin_amdgcn_wmma_f32_16x16x32_f16(false, A2, false, load_bfrag(W2p, 2*8+nt, lane), (short)0, acc2[nt].v, false, false);
      acc2[nt].v = __builtin_amdgcn_wmma_f32_16x16x32_f16(false, A3, false, load_bfrag(W2p, 3*8+nt, lane), (short)0, acc2[nt].v, false, false);
    }
    __syncthreads();
#pragma unroll
    for (int nt = 0; nt < 8; ++nt)
#pragma unroll
      for (int v = 0; v < 8; ++v)
        s_feat[(wave * 16 + v + mhi) * FSTRIDE + nt*16 + col] =
            (_Float16)fmaxf(acc2[nt].e[v], 0.0f);
    __syncthreads();

    // ---- layer 3: [16x128] x [128x16] (cols 0..2 valid) ----
    A0 = load_afrag(f, 0, lane);
    A1 = load_afrag(f, 1, lane);
    A2 = load_afrag(f, 2, lane);
    A3 = load_afrag(f, 3, lane);
    F8 o3;
    float bb3 = (col < 3) ? b3[col] : 0.0f;
#pragma unroll
    for (int v = 0; v < 8; ++v) o3.e[v] = bb3;
    o3.v = __builtin_amdgcn_wmma_f32_16x16x32_f16(false, A0, false, load_bfrag(W3p, 0, lane), (short)0, o3.v, false, false);
    o3.v = __builtin_amdgcn_wmma_f32_16x16x32_f16(false, A1, false, load_bfrag(W3p, 1, lane), (short)0, o3.v, false, false);
    o3.v = __builtin_amdgcn_wmma_f32_16x16x32_f16(false, A2, false, load_bfrag(W3p, 2, lane), (short)0, o3.v, false, false);
    o3.v = __builtin_amdgcn_wmma_f32_16x16x32_f16(false, A3, false, load_bfrag(W3p, 3, lane), (short)0, o3.v, false, false);

    // ---- composite: rgb += weight * sigmoid(o3) ----
    if (col < 3) {
      float part = 0.0f;
#pragma unroll
      for (int v = 0; v < 8; ++v) {
        float val = 1.0f / (1.0f + __expf(-o3.e[v]));
        part += s_w[base + v + mhi] * val;
      }
      atomicAdd(&s_rgb[col], part);
    }
    __syncthreads();
  }

  if (tid < 3) {
    float v = s_rgb[tid] + (1.0f - *s_acc);   // white background
    out[ray*3 + tid] = fminf(fmaxf(v, 0.0f), 1.0f);
  }
}

extern "C" void kernel_launch(void* const* d_in, const int* in_sizes, int n_in,
                              void* d_out, int out_size, void* d_ws, size_t ws_size,
                              hipStream_t stream) {
  const float* rays_o = (const float*)d_in[0];
  const float* rays_d = (const float*)d_in[1];
  const float* dens   = (const float*)d_in[2];
  const float* app    = (const float*)d_in[3];
  const float* W1     = (const float*)d_in[4];
  const float* b1     = (const float*)d_in[5];
  const float* W2     = (const float*)d_in[6];
  const float* b2     = (const float*)d_in[7];
  const float* W3     = (const float*)d_in[8];
  const float* b3     = (const float*)d_in[9];
  float* out = (float*)d_out;
  _Float16* pack = (_Float16*)d_ws;   // 68 frags * 512 halves * 2B = 68 KB

  const int N = in_sizes[0] / 3;      // 4096 rays

  // allow >64KB dynamic LDS (WGP has 320KB)
  (void)hipFuncSetAttribute(reinterpret_cast<const void*>(render_kernel),
                            hipFuncAttributeMaxDynamicSharedMemorySize,
                            SMEM_BYTES);

  wconvert<<<68, 512, 0, stream>>>(W1, W2, W3, pack);
  render_kernel<<<N, 256, SMEM_BYTES, stream>>>(rays_o, rays_d, dens, app,
                                                b1, b2, b3, pack, out);
}
